// DDIMDepthEstimateRes_79396765433973
// MI455X (gfx1250) — matrix operand order repaired and checked by reading, not compiled, not run on hardware
//
#include <hip/hip_runtime.h>
#include <math.h>

// Problem constants (match reference)
#define Bn    4
#define Cn    96
#define Hn    96
#define Wn    192
#define HWn   (Hn*Wn)            // 18432 pixels per (batch, channel)
#define NTOT  (Bn*Cn*HWn)        // 7,077,888 elements per tensor
#define NSTEP 20
#define NEVAL 21                 // 20 DDIM steps + 1 training-branch eval
#define NGRP  4
#define GSZ   (Cn/NGRP)          // 24
#define EPSV  1e-5f

typedef __attribute__((ext_vector_type(16))) __bf16 v16bf;
typedef __attribute__((ext_vector_type(8)))  float  v8f;

// ---------- helpers ----------
__device__ __forceinline__ unsigned short f2bf(float f) {
  unsigned u = __float_as_uint(f);
  u += 0x7FFFu + ((u >> 16) & 1u);      // round-to-nearest-even
  return (unsigned short)(u >> 16);
}
__device__ __forceinline__ float bf2f(unsigned short h) {
  return __uint_as_float(((unsigned)h) << 16);
}
// 16-bit A-matrix 16x32 element map: for lane L,
// vector element i -> K index within the 32-wide K tile.
// (lanes 0-15 hold K=0..15, lanes 16-31 hold K=16..31, matching bkmap)
__device__ __forceinline__ int akmap(int i, int L) {
  return (L >= 16 ? 16 : 0) + i;
}
// 16-bit B-matrix 32x16: lanes 0-15 hold K=0..15 (element i -> K=i),
// lanes 16-31 hold K=16..31.
__device__ __forceinline__ int bkmap(int i, int L) {
  return (L >= 16 ? 16 : 0) + i;
}
__device__ __forceinline__ v8f vzero8() {
  v8f z;
#pragma unroll
  for (int q = 0; q < 8; ++q) z[q] = 0.f;
  return z;
}
__device__ __forceinline__ v8f wmma_bf(v16bf a, v16bf b, v8f c) {
  return __builtin_amdgcn_wmma_f32_16x16x32_bf16(false, a, false, b, (short)0, c,
                                                 false, false);
}

struct TSList { int t[NSTEP]; };

// ---------- prep: repack W1/W2 into WMMA A-fragment layout, zero accumulators ----------
__global__ void prep_kernel(const float* __restrict__ w1, const float* __restrict__ w2,
                            unsigned short* __restrict__ w1b, unsigned short* __restrict__ w2b,
                            float* __restrict__ Sfp, float* __restrict__ fpsum,
                            float* __restrict__ st2) {
  int tid = blockIdx.x * blockDim.x + threadIdx.x;
  int nth = gridDim.x * blockDim.x;
  // 6 m-tiles x 3 k-tiles x 32 lanes x 16 halves
  for (int o = tid; o < 6 * 3 * 32 * 16; o += nth) {
    int i = o & 15;
    int L = (o >> 4) & 31;
    int s = o >> 9;                  // mt*3 + kt
    int kt = s % 3, mt = s / 3;
    int m = mt * 16 + (L & 15);      // A row = lane%16
    int k = kt * 32 + akmap(i, L);
    w1b[o] = f2bf(w1[m * Cn + k]);
    w2b[o] = f2bf(w2[m * Cn + k]);
  }
  for (int o = tid; o < Bn * Cn * Cn; o += nth) Sfp[o] = 0.f;
  for (int o = tid; o < Bn * Cn; o += nth) fpsum[o] = 0.f;
  for (int o = tid; o < NEVAL * Bn * NGRP * 2; o += nth) st2[o] = 0.f;
}

// ---------- init outputs: refined = (prod alpha) * init_image ; noise passthrough ----------
__global__ void init_out_kernel(const float4* __restrict__ init_image,
                                const float4* __restrict__ noise,
                                float4* __restrict__ out, float pa) {
  int i = blockIdx.x * 256 + threadIdx.x;
  if (i < NTOT / 4) {
    float4 v = init_image[i];
    float4 o; o.x = pa * v.x; o.y = pa * v.y; o.z = pa * v.z; o.w = pa * v.w;
    out[i] = o;
    out[2 * (NTOT / 4) + i] = noise[i];
  }
}

// ---------- per-batch moments of fp: fpsum[b][c] and Sfp[b][96][96] = sum_p x x^T ----------
// S is computed with bf16 WMMA: A = X tile (ch x px), B = X^T tile (px x ch).
#define MSEG   12              // segments per batch
#define MCHUNK 6               // 256-pixel chunks per segment (12*6*256 = 18432)
__global__ __launch_bounds__(256) void moments_kernel(const float* __restrict__ fp,
                                                      float* __restrict__ Sfp,
                                                      float* __restrict__ fpsum) {
  __shared__ unsigned short Xs[Cn][256 + 8];
  int blk = blockIdx.x;                // Bn*MSEG blocks
  int b = blk / MSEG, seg = blk % MSEG;
  int tid = threadIdx.x;
  int wave = tid >> 5, L = tid & 31;
  v8f acc[5];
#pragma unroll
  for (int ti = 0; ti < 5; ++ti) acc[ti] = vzero8();
  float rowsum = 0.f;

  for (int ch = 0; ch < MCHUNK; ++ch) {
    int p0 = (seg * MCHUNK + ch) * 256;
    __syncthreads();
    for (int idx = tid; idx < Cn * 64; idx += 256) {   // float4 loads
      int c = idx >> 6, p4 = idx & 63;
      float4 v = *(const float4*)(fp + (b * Cn + c) * HWn + p0 + p4 * 4);
      Xs[c][p4 * 4 + 0] = f2bf(v.x);
      Xs[c][p4 * 4 + 1] = f2bf(v.y);
      Xs[c][p4 * 4 + 2] = f2bf(v.z);
      Xs[c][p4 * 4 + 3] = f2bf(v.w);
    }
    __syncthreads();
    if (tid < Cn) {
      float s = 0.f;
      for (int p = 0; p < 256; ++p) s += bf2f(Xs[tid][p]);
      rowsum += s;
    }
#pragma unroll
    for (int ti = 0; ti < 5; ++ti) {
      int t = wave + ti * 8;           // wave-uniform mask: EXEC stays full
      if (t < 36) {
        int mt = t / 6, nt = t % 6;
        for (int kc = 0; kc < 8; ++kc) {
          v16bf a, bb;
#pragma unroll
          for (int i = 0; i < 16; ++i) {
            a[i]  = __builtin_bit_cast(__bf16,
                     Xs[mt * 16 + (L & 15)][kc * 32 + akmap(i, L)]);
            bb[i] = __builtin_bit_cast(__bf16,
                     Xs[nt * 16 + (L & 15)][kc * 32 + bkmap(i, L)]);
          }
          acc[ti] = wmma_bf(a, bb, acc[ti]);
        }
      }
    }
  }
  if (tid < Cn) atomicAdd(&fpsum[b * Cn + tid], rowsum);
#pragma unroll
  for (int ti = 0; ti < 5; ++ti) {
    int t = wave + ti * 8;
    if (t < 36) {
      int mt = t / 6, nt = t % 6;
#pragma unroll
      for (int j = 0; j < 8; ++j) {
        int m = mt * 16 + j + (L >= 16 ? 8 : 0);
        int n = nt * 16 + (L & 15);
        atomicAdd(&Sfp[(b * Cn + m) * Cn + n], acc[ti][j]);
      }
    }
  }
}

// ---------- closed-form GroupNorm-1 stats per (eval, batch) ----------
// h_c(p) = w_c . x_p + b1_c,  x = fp + e.  Uses fpsum / Sfp moments.
__global__ __launch_bounds__(96) void stats1_kernel(
    const float* __restrict__ w1, const float* __restrict__ b1,
    const float* __restrict__ g1w, const float* __restrict__ g1b,
    const float* __restrict__ emb, const int* __restrict__ tsteps_train,
    const float* __restrict__ fpsum, const float* __restrict__ Sfp,
    float* __restrict__ st1, TSList tsl) {
  __shared__ float smu[Cn], se2[Cn], ginv[NGRP], gmu[NGRP];
  int e = blockIdx.x / Bn, b = blockIdx.x % Bn;
  int c = threadIdx.x;
  int t = (e < NSTEP) ? tsl.t[e] : tsteps_train[b];
  const float* wr = w1 + c * Cn;
  const float* er = emb + t * Cn;
  const float* fr = fpsum + b * Cn;
  float we = 0.f, wf = 0.f;
  for (int k = 0; k < Cn; ++k) { we += wr[k] * er[k]; wf += wr[k] * fr[k]; }
  float q = 0.f;                       // w^T Sfp w
  for (int j = 0; j < Cn; ++j) {
    const float* sr = Sfp + (b * Cn + j) * Cn;
    float inner = 0.f;
    for (int k = 0; k < Cn; ++k) inner += sr[k] * wr[k];
    q += wr[j] * inner;
  }
  const float invNp = 1.f / (float)HWn;
  float b1c = b1[c];
  float mu  = wf * invNp + we + b1c;                  // E_p[h_c]
  float Ewx = mu - b1c;
  float Ewx2 = (q + 2.f * wf * we) * invNp + we * we; // E_p[(w.x)^2]
  float e2 = Ewx2 + 2.f * b1c * Ewx + b1c * b1c;      // E_p[h_c^2]
  smu[c] = mu; se2[c] = e2;
  __syncthreads();
  if (c < NGRP) {
    float sm = 0.f, s2 = 0.f;
    for (int k = 0; k < GSZ; ++k) { sm += smu[c * GSZ + k]; s2 += se2[c * GSZ + k]; }
    sm *= (1.f / GSZ); s2 *= (1.f / GSZ);
    ginv[c] = rsqrtf(s2 - sm * sm + EPSV);
    gmu[c]  = sm;
  }
  __syncthreads();
  int g = c / GSZ;
  float s  = ginv[g] * g1w[c];
  float tt = (b1c - gmu[g]) * s + g1b[c];             // bias b1 folded in
  st1[((e * Bn + b) * Cn + c) * 2 + 0] = s;
  st1[((e * Bn + b) * Cn + c) * 2 + 1] = tt;
}

// ---------- fused GEMM1 -> GN1 -> ReLU -> GEMM2 (+ GN2 partial sums) ----------
// One workgroup = 8 waves, 128 pixels. LDS holds activations in B-fragment-major
// layout so each lane's v16bf fragment is one contiguous 32B LDS load.
__global__ __launch_bounds__(256) void fused_kernel(
    const float* __restrict__ fp, const float* __restrict__ emb,
    const int* __restrict__ tsteps_train,
    const unsigned short* __restrict__ w1b, const unsigned short* __restrict__ w2b,
    const float* __restrict__ b2, const float* __restrict__ st1,
    float* __restrict__ st2, unsigned short* __restrict__ h2buf,
    int e, int t_fixed) {
  __shared__ __align__(32) unsigned short Xf [8 * 3 * 32 * 16];
  __shared__ __align__(32) unsigned short H1f[8 * 3 * 32 * 16];
  __shared__ float s1s[Cn], s1t[Cn], sb2[Cn], semb[Cn];
  __shared__ float gsum[NGRP], gsq[NGRP];

  int blk = blockIdx.x;
  int b = blk / (HWn / 128), tile = blk % (HWn / 128);
  int p0 = tile * 128;
  int tid = threadIdx.x;
  int t = (t_fixed >= 0) ? t_fixed : tsteps_train[b];

  if (tid < Cn) {
    s1s[tid]  = st1[((e * Bn + b) * Cn + tid) * 2 + 0];
    s1t[tid]  = st1[((e * Bn + b) * Cn + tid) * 2 + 1];
    sb2[tid]  = b2[tid];
    semb[tid] = emb[t * Cn + tid];
  }
  if (tid < NGRP) { gsum[tid] = 0.f; gsq[tid] = 0.f; }
  __syncthreads();

  // Load fp tile (float4) + embedding, scatter into B-fragment layout.
  // p, p+1, p+2, p+3 map to consecutive lanes -> LDS stride of 16 halves.
  for (int idx = tid; idx < Cn * 32; idx += 256) {
    int c = idx >> 5, p4 = idx & 31;
    float4 v = *(const float4*)(fp + (b * Cn + c) * HWn + p0 + p4 * 4);
    float ec = semb[c];
    int kt = c >> 5, kk = c & 31;
    int i  = kk & 15;
    int hh = (kk >= 16) ? 16 : 0;
    int p  = p4 * 4;
    int wv = p >> 4;
    int L0 = (p & 15) + hh;
    int base = ((wv * 3 + kt) * 32 + L0) * 16 + i;
    Xf[base +  0] = f2bf(v.x + ec);
    Xf[base + 16] = f2bf(v.y + ec);
    Xf[base + 32] = f2bf(v.z + ec);
    Xf[base + 48] = f2bf(v.w + ec);
  }
  __syncthreads();

  int wave = tid >> 5, L = tid & 31;
  int col = L & 15;
  int off = (L >= 16) ? 8 : 0;
  bool hi = (L >= 16);
  const v16bf* w1f = (const v16bf*)w1b;
  const v16bf* w2f = (const v16bf*)w2b;

  // GEMM1: 6 m-tiles x (K=96 as 3 x 32)
  v16bf bx0 = *(const v16bf*)&Xf[((wave * 3 + 0) * 32 + L) * 16];
  v16bf bx1 = *(const v16bf*)&Xf[((wave * 3 + 1) * 32 + L) * 16];
  v16bf bx2 = *(const v16bf*)&Xf[((wave * 3 + 2) * 32 + L) * 16];
#pragma unroll
  for (int mt = 0; mt < 6; ++mt) {
    v8f acc = vzero8();
    acc = wmma_bf(w1f[(mt * 3 + 0) * 32 + L], bx0, acc);
    acc = wmma_bf(w1f[(mt * 3 + 1) * 32 + L], bx1, acc);
    acc = wmma_bf(w1f[(mt * 3 + 2) * 32 + L], bx2, acc);
#pragma unroll
    for (int j = 0; j < 8; ++j) {
      int m = mt * 16 + j + off;
      float y = fmaxf(acc[j] * s1s[m] + s1t[m], 0.f);   // GN1 + ReLU
      int kt2 = m >> 5, kk = m & 31;
      int L2 = col + (kk >= 16 ? 16 : 0);
      H1f[((wave * 3 + kt2) * 32 + L2) * 16 + (kk & 15)] = f2bf(y);
    }
  }
  __syncthreads();

  // GEMM2 + GN2 partial sums + store h2 (bf16)
  v16bf bh0 = *(const v16bf*)&H1f[((wave * 3 + 0) * 32 + L) * 16];
  v16bf bh1 = *(const v16bf*)&H1f[((wave * 3 + 1) * 32 + L) * 16];
  v16bf bh2 = *(const v16bf*)&H1f[((wave * 3 + 2) * 32 + L) * 16];
  float ps[NGRP] = {0.f, 0.f, 0.f, 0.f}, pq[NGRP] = {0.f, 0.f, 0.f, 0.f};
#pragma unroll
  for (int mt = 0; mt < 6; ++mt) {
    v8f acc = vzero8();
    acc = wmma_bf(w2f[(mt * 3 + 0) * 32 + L], bh0, acc);
    acc = wmma_bf(w2f[(mt * 3 + 1) * 32 + L], bh1, acc);
    acc = wmma_bf(w2f[(mt * 3 + 2) * 32 + L], bh2, acc);
#pragma unroll
    for (int j = 0; j < 8; ++j) {
      const int mm = mt * 16 + j;       // compile-time after unroll
      const int g0 = mm / GSZ;          // group when off == 0
      const int g1 = (mm + 8) / GSZ;    // group when off == 8
      int m = mm + off;
      float h = acc[j] + sb2[m];
      h2buf[(b * Cn + m) * HWn + p0 + wave * 16 + col] = f2bf(h);
      // Constant-indexed group accumulation; only mt=1 / mt=4 straddle a
      // group boundary, and the pick there is half-wave-uniform (hi).
      if (g0 == g1) { ps[g0] += h; pq[g0] += h * h; }
      else if (hi)  { ps[g1] += h; pq[g1] += h * h; }
      else          { ps[g0] += h; pq[g0] += h * h; }
    }
  }
#pragma unroll
  for (int gg = 0; gg < NGRP; ++gg) {
    atomicAdd(&gsum[gg], ps[gg]);      // ds_add_f32
    atomicAdd(&gsq[gg],  pq[gg]);
  }
  __syncthreads();
  if (tid < NGRP) {
    atomicAdd(&st2[((e * Bn + b) * NGRP + tid) * 2 + 0], gsum[tid]);
    atomicAdd(&st2[((e * Bn + b) * NGRP + tid) * 2 + 1], gsq[tid]);
  }
}

// ---------- GN2 + accumulate c_k * mo into refined (or write noise_pred) ----------
// 4 elements per thread; (b, c, g) constant within a quad since HWn % 4 == 0.
__global__ void finalize_kernel(const unsigned short* __restrict__ h2buf,
                                const float* __restrict__ st2,
                                const float* __restrict__ g2w,
                                const float* __restrict__ g2b,
                                float* __restrict__ out, int e, float ck) {
  int i4 = blockIdx.x * 256 + threadIdx.x;
  if (i4 >= NTOT / 4) return;
  int i = i4 * 4;
  int c = (i / HWn) % Cn;
  int b = i / (Cn * HWn);
  int g = c / GSZ;
  float sum = st2[((e * Bn + b) * NGRP + g) * 2 + 0];
  float sq  = st2[((e * Bn + b) * NGRP + g) * 2 + 1];
  const float invN = 1.f / (float)(GSZ * HWn);
  float mu  = sum * invN;
  float var = sq * invN - mu * mu;
  float inv = rsqrtf(var + EPSV);
  float sc = inv * g2w[c];
  float sh = g2b[c] - mu * sc;
  ushort4 hv = *(const ushort4*)(h2buf + i);
  float4 y;
  y.x = bf2f(hv.x) * sc + sh;
  y.y = bf2f(hv.y) * sc + sh;
  y.z = bf2f(hv.z) * sc + sh;
  y.w = bf2f(hv.w) * sc + sh;
  if (e < NSTEP) {
    float4 cur = *(float4*)(out + i);
    cur.x += ck * y.x; cur.y += ck * y.y; cur.z += ck * y.z; cur.w += ck * y.w;
    *(float4*)(out + i) = cur;
  } else {
    *(float4*)(out + NTOT + i) = y;
  }
}

// ---------- host ----------
extern "C" void kernel_launch(void* const* d_in, const int* in_sizes, int n_in,
                              void* d_out, int out_size, void* d_ws, size_t ws_size,
                              hipStream_t stream) {
  const float* fp         = (const float*)d_in[0];
  const float* init_image = (const float*)d_in[3];
  const float* noise      = (const float*)d_in[4];
  const int*   tst        = (const int*)d_in[5];
  const float* emb        = (const float*)d_in[6];
  const float* w1  = (const float*)d_in[7];
  const float* b1  = (const float*)d_in[8];
  const float* g1w = (const float*)d_in[9];
  const float* g1b = (const float*)d_in[10];
  const float* w2  = (const float*)d_in[11];
  const float* b2  = (const float*)d_in[12];
  const float* g2w = (const float*)d_in[13];
  const float* g2b = (const float*)d_in[14];
  float* out = (float*)d_out;

  // workspace carve (~14.5 MB total)
  char* ws = (char*)d_ws;
  size_t off = 0;
  auto alloc = [&](size_t bytes) -> void* {
    off = (off + 255) & ~(size_t)255;
    void* p = ws + off; off += bytes; return p;
  };
  unsigned short* w1b   = (unsigned short*)alloc(6 * 3 * 32 * 16 * sizeof(unsigned short));
  unsigned short* w2b   = (unsigned short*)alloc(6 * 3 * 32 * 16 * sizeof(unsigned short));
  float*          fpsum = (float*)alloc(Bn * Cn * sizeof(float));
  float*          Sfp   = (float*)alloc((size_t)Bn * Cn * Cn * sizeof(float));
  float*          st1   = (float*)alloc((size_t)NEVAL * Bn * Cn * 2 * sizeof(float));
  float*          st2   = (float*)alloc((size_t)NEVAL * Bn * NGRP * 2 * sizeof(float));
  unsigned short* h2buf = (unsigned short*)alloc((size_t)NTOT * sizeof(unsigned short));

  // DDIM constants (diffusers DDIM defaults), host-side double precision
  double acp[1000], prod = 1.0;
  for (int i = 0; i < 1000; ++i) {
    double bta = 1e-4 + (0.02 - 1e-4) * (double)i / 999.0;
    prod *= (1.0 - bta);
    acp[i] = prod;
  }
  double alpha[NSTEP], beta[NSTEP];
  TSList tsl;
  for (int k = 0; k < NSTEP; ++k) {
    int t = (19 - k) * 50;             // ts = [950, 900, ..., 0]
    tsl.t[k] = t;
    double a_t = acp[t];
    double a_p = (t - 50 >= 0) ? acp[t - 50] : 1.0;
    double sa_t = sqrt(a_t), sb_t = sqrt(1.0 - a_t);
    double sa_p = sqrt(a_p), sb_p = sqrt(1.0 - a_p);
    alpha[k] = sa_p / sa_t;
    beta[k]  = sb_p - sa_p * sb_t / sa_t;
  }
  double pa = 1.0;
  for (int k = 0; k < NSTEP; ++k) pa *= alpha[k];
  double ckd[NSTEP];
  for (int k = 0; k < NSTEP; ++k) {
    double pr = 1.0;
    for (int j = k + 1; j < NSTEP; ++j) pr *= alpha[j];
    ckd[k] = beta[k] * pr;
  }

  const int nblk4 = (NTOT / 4 + 255) / 256;
  prep_kernel<<<160, 256, 0, stream>>>(w1, w2, w1b, w2b, Sfp, fpsum, st2);
  init_out_kernel<<<nblk4, 256, 0, stream>>>((const float4*)init_image,
                                             (const float4*)noise,
                                             (float4*)out, (float)pa);
  moments_kernel<<<Bn * MSEG, 256, 0, stream>>>(fp, Sfp, fpsum);
  stats1_kernel<<<NEVAL * Bn, 96, 0, stream>>>(w1, b1, g1w, g1b, emb, tst,
                                               fpsum, Sfp, st1, tsl);
  for (int e = 0; e < NEVAL; ++e) {
    int tf = (e < NSTEP) ? tsl.t[e] : -1;
    fused_kernel<<<Bn * (HWn / 128), 256, 0, stream>>>(fp, emb, tst, w1b, w2b, b2,
                                                       st1, st2, h2buf, e, tf);
    float ck = (e < NSTEP) ? (float)ckd[e] : 0.f;
    finalize_kernel<<<nblk4, 256, 0, stream>>>(h2buf, st2, g2w, g2b, out, e, ck);
  }
}
